// RNN_SRU_69200513073539
// MI455X (gfx1250) — matrix operand order repaired and checked by reading, not compiled, not run on hardware
//
#include <hip/hip_runtime.h>
#include <hip/hip_bf16.h>

// ---------------------------------------------------------------------------
// Problem constants (from reference): T=512, B=32, NIN=72, NOUT=90, H=512, L=5
// ---------------------------------------------------------------------------
#define TT   512
#define BB   32
#define NIN  72
#define NOUT 90
#define HH   512
#define MM   (TT * BB)          // 16384 rows for all GEMMs

typedef __bf16 v16bf __attribute__((ext_vector_type(16)));
typedef __bf16 v8bf  __attribute__((ext_vector_type(8)));
typedef float  v8f   __attribute__((ext_vector_type(8)));

union AFrag { v8bf h[2]; v16bf v; };

// ---------------------------------------------------------------------------
// Fast device math
// ---------------------------------------------------------------------------
__device__ __forceinline__ float sigmoidf_(float x) {
    return 1.0f / (1.0f + __expf(-x));
}
__device__ __forceinline__ float tanhf_(float x) {
    float a = fabsf(x);
    float e = __expf(-2.0f * a);
    float t = (1.0f - e) / (1.0f + e);
    return x < 0.0f ? -t : t;
}

// ---------------------------------------------------------------------------
// Weight pack: fp32 (K x Nsrc) row-major  ->  bf16 packed B-fragments.
// Per 32-K x 16-N tile, lane l holds 16 bf16 along K:
//   kr in [0,16): lane = n%16,        e = kr
//   kr in [16,32): lane = 16 + n%16,  e = kr-16
// Packed index: (((nt*KT + kt)*32 + lane)*16 + e).  n >= Nsrc is zero padding.
// ---------------------------------------------------------------------------
__global__ void k_packB(const float* __restrict__ W, __bf16* __restrict__ P,
                        int K, int Npad, int Nsrc) {
    size_t i = (size_t)blockIdx.x * blockDim.x + threadIdx.x;
    size_t total = (size_t)K * Npad;
    if (i >= total) return;
    int k = (int)(i / Npad);
    int n = (int)(i % Npad);
    float w = (n < Nsrc) ? W[(size_t)k * Nsrc + n] : 0.0f;
    int kt = k >> 5, kr = k & 31;
    int lane = (n & 15) + ((kr >> 4) << 4);
    int e    = kr & 15;
    int nt   = n >> 4;
    int KT   = K >> 5;
    size_t p = ((((size_t)nt * KT + kt) * 32 + lane) << 4) + e;
    P[p] = (__bf16)w;
}

// ---------------------------------------------------------------------------
// Input projection: h0 = relu(x @ W1 + b1).   x:(M,72) W1:(72,512)
// One block per row; x row staged in LDS; W1 (144 KB) lives in L2.
// ---------------------------------------------------------------------------
__global__ void k_inproj(const float* __restrict__ x, const float* __restrict__ W1,
                         const float* __restrict__ b1, float* __restrict__ h0) {
    __shared__ float xr[NIN];
    size_t row = blockIdx.x;
    if (threadIdx.x < NIN) xr[threadIdx.x] = x[row * NIN + threadIdx.x];
    __syncthreads();
    for (int n = threadIdx.x; n < HH; n += blockDim.x) {
        float acc = b1[n];
        #pragma unroll 8
        for (int k = 0; k < NIN; ++k) acc = fmaf(xr[k], W1[(size_t)k * HH + n], acc);
        h0[row * HH + n] = fmaxf(acc, 0.0f);
    }
}

// ---------------------------------------------------------------------------
// LayerNorm over feature dim D + cast to bf16. One block (256 thr) per row.
// ---------------------------------------------------------------------------
__global__ void k_ln_cast(const float* __restrict__ X, const float* __restrict__ g,
                          const float* __restrict__ be, __bf16* __restrict__ Y, int D) {
    __shared__ float s1[256];
    __shared__ float s2[256];
    size_t row = blockIdx.x;
    const float* xr = X + row * (size_t)D;
    float a = 0.0f, b = 0.0f;
    for (int i = threadIdx.x; i < D; i += 256) {
        float v = xr[i];
        a += v; b += v * v;
    }
    s1[threadIdx.x] = a; s2[threadIdx.x] = b;
    __syncthreads();
    for (int s = 128; s > 0; s >>= 1) {
        if (threadIdx.x < (unsigned)s) {
            s1[threadIdx.x] += s1[threadIdx.x + s];
            s2[threadIdx.x] += s2[threadIdx.x + s];
        }
        __syncthreads();
    }
    float invD = 1.0f / (float)D;
    float mean = s1[0] * invD;
    float var  = s2[0] * invD - mean * mean;
    float r    = rsqrtf(var + 1e-5f);
    for (int i = threadIdx.x; i < D; i += 256)
        Y[row * D + i] = (__bf16)((xr[i] - mean) * r * g[i] + be[i]);
}

// Plain fp32 -> bf16 cast (no LN) for the output projection A-matrix.
__global__ void k_cast(const float* __restrict__ X, __bf16* __restrict__ Y, size_t n) {
    size_t i = (size_t)blockIdx.x * blockDim.x + threadIdx.x;
    if (i < n) Y[i] = (__bf16)X[i];
}

// ---------------------------------------------------------------------------
// bf16 WMMA GEMM:  C(MxN,f32) = A(MxK,bf16 row-major) @ Bpacked(KxN,bf16)
// 4 waves / block; each wave computes a 64x64 tile (16 accumulators).
// Requires: M % 256 == 0, N % 64 == 0, K % 32 == 0.
// ---------------------------------------------------------------------------
__global__ __launch_bounds__(128) void
k_gemm(const __bf16* __restrict__ A, const __bf16* __restrict__ Bp,
       float* __restrict__ C, int M, int N, int K) {
    const int lane = threadIdx.x & 31;
    const int w    = threadIdx.x >> 5;
    const int m0   = blockIdx.y * 256 + w * 64;
    const int n0   = blockIdx.x * 64;
    const int KT   = K >> 5;

    v8f acc[4][4];
    #pragma unroll
    for (int mi = 0; mi < 4; ++mi)
        #pragma unroll
        for (int ni = 0; ni < 4; ++ni) acc[mi][ni] = (v8f){};

    const int mr = lane & 15;           // A row within 16-row tile
    const int kh = (lane >> 4) << 4;    // K sub-offset: 0 or 16
    const int nt0 = n0 >> 4;

    for (int kt = 0; kt < KT; ++kt) {
        v16bf a[4], b[4];
        #pragma unroll
        for (int mi = 0; mi < 4; ++mi) {
            const v8bf* pa = (const v8bf*)(A + (size_t)(m0 + mi * 16 + mr) * K
                                             + ((size_t)kt << 5) + kh);
            AFrag f;
            f.h[0] = pa[0];   // K: kh .. kh+7
            f.h[1] = pa[1];   // K: kh+8 .. kh+15
            a[mi] = f.v;
        }
        #pragma unroll
        for (int ni = 0; ni < 4; ++ni) {
            const v16bf* pb = (const v16bf*)(Bp +
                (((((size_t)(nt0 + ni)) * KT + kt) * 32 + lane) << 4));
            b[ni] = *pb;
        }
        #pragma unroll
        for (int mi = 0; mi < 4; ++mi)
            #pragma unroll
            for (int ni = 0; ni < 4; ++ni)
                acc[mi][ni] = __builtin_amdgcn_wmma_f32_16x16x32_bf16(
                    false, a[mi], false, b[ni], (short)0, acc[mi][ni], false, false);
    }

    // C/D layout: VGPR v, lane l -> row = v + 8*(l>=16), col = l%16
    const int rh = (lane >> 4) << 3;
    const int cc = lane & 15;
    #pragma unroll
    for (int mi = 0; mi < 4; ++mi)
        #pragma unroll
        for (int ni = 0; ni < 4; ++ni) {
            float* cp = C + (size_t)(m0 + mi * 16 + rh) * N + (n0 + ni * 16 + cc);
            #pragma unroll
            for (int v = 0; v < 8; ++v)
                cp[(size_t)v * N] = acc[mi][ni][v];
        }
}

// ---------------------------------------------------------------------------
// Bidirectional SRU scan. One thread per (dir, b, j).  U row width Nw = 2*k*H.
// dir 0 iterates t = 0..T-1, dir 1 iterates t = T-1..0; all indexing stays in
// original time coordinates (equivalent to the reference's flip/unflip).
// hw = U component 3 when k==4, else the layer input hin[t].
// ---------------------------------------------------------------------------
__global__ void k_scan(const float* __restrict__ U, const float* __restrict__ hin,
                       const float* __restrict__ v, const float* __restrict__ bbv,
                       float* __restrict__ hout, float* __restrict__ cout,
                       int k, int Nw) {
    int tid = blockIdx.x * blockDim.x + threadIdx.x;   // < 2*B*H
    int d = tid / (BB * HH);
    int r = tid % (BB * HH);
    int b = r / HH;
    int j = r % HH;

    const float vf = v[(d * 2 + 0) * HH + j];
    const float vr = v[(d * 2 + 1) * HH + j];
    const float bf = bbv[(d * 2 + 0) * HH + j];
    const float br = bbv[(d * 2 + 1) * HH + j];

    const int t0 = d ? (TT - 1) : 0;
    const int dt = d ? -1 : 1;
    const size_t colbase = (size_t)d * k * HH + j;

    float c = 0.0f;
    for (int s = 0; s < TT; ++s) {
        int t = t0 + s * dt;
        const float* u = U + ((size_t)t * BB + b) * Nw + colbase;
        float xt = u[0];
        float ft = u[HH];
        float rt = u[2 * HH];
        float hw = (k == 4) ? u[3 * HH]
                            : hin[((size_t)t * BB + b) * HH + j];
        float f = sigmoidf_(ft + vf * c + bf);
        float g = sigmoidf_(rt + vr * c + br);
        c = f * c + (1.0f - f) * xt;
        float h = g * tanhf_(c) + (1.0f - g) * hw;
        hout[((size_t)t * BB + b) * (2 * HH) + (size_t)d * HH + j] = h;
    }
    cout[(size_t)b * (2 * HH) + (size_t)d * HH + j] = c;
}

// ---------------------------------------------------------------------------
// Epilogue: out[m, 0..89] = U[m, 0..89 (of 128 padded)] + b2
// ---------------------------------------------------------------------------
__global__ void k_outbias(const float* __restrict__ U, const float* __restrict__ b2,
                          float* __restrict__ out) {
    size_t i = (size_t)blockIdx.x * blockDim.x + threadIdx.x;
    if (i >= (size_t)MM * NOUT) return;
    size_t m = i / NOUT;
    int n = (int)(i % NOUT);
    out[i] = U[m * 128 + n] + b2[n];
}

// ---------------------------------------------------------------------------
// Host-side orchestration (stream-ordered, graph-capture safe).
// ---------------------------------------------------------------------------
extern "C" void kernel_launch(void* const* d_in, const int* in_sizes, int n_in,
                              void* d_out, int out_size, void* d_ws, size_t ws_size,
                              hipStream_t stream) {
    (void)in_sizes; (void)n_in; (void)out_size; (void)ws_size;

    const float* x   = (const float*)d_in[0];
    const float* W1  = (const float*)d_in[1];
    const float* b1  = (const float*)d_in[2];
    const float* W2  = (const float*)d_in[3];
    const float* b2  = (const float*)d_in[4];
    const float* W0  = (const float*)d_in[5];
    const float* v0  = (const float*)d_in[6];
    const float* bb0 = (const float*)d_in[7];
    const float* g0  = (const float*)d_in[8];
    const float* be0 = (const float*)d_in[9];
    const float* Ws  = (const float*)d_in[10];
    const float* vs  = (const float*)d_in[11];
    const float* bbs = (const float*)d_in[12];
    const float* gs  = (const float*)d_in[13];
    const float* bes = (const float*)d_in[14];

    float* out    = (float*)d_out;
    float* states = out + (size_t)MM * NOUT;   // (5, B, 2H)

    // Workspace carve-up (~453 MB)
    char* ws = (char*)d_ws;
    float*  hA  = (float*)(ws);                         //  64 MB ping
    float*  hB  = (float*)(ws + (64ull  << 20));        //  64 MB pong
    __bf16* xnb = (__bf16*)(ws + (128ull << 20));       //  32 MB bf16 A-matrix
    float*  U   = (float*)(ws + (160ull << 20));        // 256 MB GEMM output
    __bf16* P0  = (__bf16*)(ws + (416ull << 20));       //   3 MB packed W0
    __bf16* P1  = (__bf16*)(ws + (420ull << 20));       //   8 MB each
    __bf16* P2  = (__bf16*)(ws + (428ull << 20));
    __bf16* P3  = (__bf16*)(ws + (436ull << 20));
    __bf16* P4  = (__bf16*)(ws + (444ull << 20));
    __bf16* P5  = (__bf16*)(ws + (452ull << 20));       // 256 KB packed W2 (N->128)
    __bf16* PL[4] = {P1, P2, P3, P4};

    // 1) Pack weights to bf16 WMMA-B layout
    {
        size_t t0 = (size_t)HH * (6 * HH);               // 512 x 3072
        k_packB<<<dim3((unsigned)((t0 + 255) / 256)), 256, 0, stream>>>(W0, P0, HH, 6 * HH, 6 * HH);
        size_t t1 = (size_t)(2 * HH) * (8 * HH);         // 1024 x 4096
        for (int i = 0; i < 4; ++i)
            k_packB<<<dim3((unsigned)((t1 + 255) / 256)), 256, 0, stream>>>(
                Ws + (size_t)i * t1, PL[i], 2 * HH, 8 * HH, 8 * HH);
        size_t t2 = (size_t)(2 * HH) * 128;              // 1024 x 128 (90 + pad)
        k_packB<<<dim3((unsigned)((t2 + 255) / 256)), 256, 0, stream>>>(W2, P5, 2 * HH, 128, NOUT);
    }

    // 2) Input projection + ReLU -> hA (M x 512)
    k_inproj<<<dim3(MM), 256, 0, stream>>>(x, W1, b1, hA);

    // 3) Layer 0 (k=3): LN(512) -> GEMM (K=512, N=3072) -> scan -> hB
    k_ln_cast<<<dim3(MM), 256, 0, stream>>>(hA, g0, be0, xnb, HH);
    k_gemm<<<dim3((6 * HH) / 64, MM / 256), 128, 0, stream>>>(xnb, P0, U, MM, 6 * HH, HH);
    k_scan<<<dim3((2 * BB * HH) / 256), 256, 0, stream>>>(U, hA, v0, bb0, hB,
                                                          states, 3, 6 * HH);

    // 4) Layers 1..4 (k=4): LN(1024) -> GEMM (K=1024, N=4096) -> scan
    float* hcur = hB;
    float* hnxt = hA;
    for (int i = 0; i < 4; ++i) {
        k_ln_cast<<<dim3(MM), 256, 0, stream>>>(hcur, gs + (size_t)i * (2 * HH),
                                                bes + (size_t)i * (2 * HH), xnb, 2 * HH);
        k_gemm<<<dim3((8 * HH) / 64, MM / 256), 128, 0, stream>>>(xnb, PL[i], U,
                                                                  MM, 8 * HH, 2 * HH);
        k_scan<<<dim3((2 * BB * HH) / 256), 256, 0, stream>>>(
            U, nullptr,
            vs + (size_t)i * (2 * 2 * HH), bbs + (size_t)i * (2 * 2 * HH),
            hnxt, states + (size_t)(i + 1) * (BB * 2 * HH), 4, 8 * HH);
        float* tmp = hcur; hcur = hnxt; hnxt = tmp;
    }

    // 5) Output projection: cast -> GEMM (K=1024, N=128 padded) -> bias+slice
    {
        size_t n = (size_t)MM * (2 * HH);
        k_cast<<<dim3((unsigned)((n + 255) / 256)), 256, 0, stream>>>(hcur, xnb, n);
        k_gemm<<<dim3(128 / 64, MM / 256), 128, 0, stream>>>(xnb, P5, U, MM, 128, 2 * HH);
        size_t no = (size_t)MM * NOUT;
        k_outbias<<<dim3((unsigned)((no + 255) / 256)), 256, 0, stream>>>(U, b2, out);
    }
}
